// CostVolumeLayer_893353197639
// MI455X (gfx1250) — matrix operand order repaired and checked by reading, not compiled, not run on hardware
//
#include <hip/hip_runtime.h>
#include <hip/hip_bf16.h>

typedef __attribute__((ext_vector_type(16))) _Float16 v16h;
typedef __attribute__((ext_vector_type(8)))  float    v8f;

#define SR   4
#define DD   9            // 2*SR+1
#define NC   81           // DD*DD
#define NB   8
#define NCH  64
#define NH   192
#define NW   256
#define WPB  8            // waves per block (wave32)

// One wave handles a 16-wide pixel tile of one (b,h) row: builds A once
// (16 pix x 64 ch, f16), then for each vertical shift i loads two B tiles
// of x2 row h-i (cols w0-4..w0+27), runs 4x v_wmma_f32_16x16x32_f16, and
// extracts the 9 diagonals (horizontal shifts) via a small LDS transpose
// so stores are coalesced 64B runs per k-plane.
__global__ __launch_bounds__(32 * WPB)
void costvol_wmma_kernel(const float* __restrict__ x1,
                         const float* __restrict__ x2,
                         float* __restrict__ out)
{
    __shared__ float lds_out[WPB][DD * 16];

    const int lane = threadIdx.x & 31;
    const int wave = threadIdx.x >> 5;
    const int half = lane >> 4;     // 0: lanes 0-15, 1: lanes 16-31
    const int sub  = lane & 15;

    const int wgroups = (NW / 16) / WPB;              // 2
    int bid = blockIdx.x;
    const int wg = bid % wgroups;
    const int h  = (bid / wgroups) % NH;
    const int bb = bid / (wgroups * NH);
    const int w0 = (wg * WPB + wave) * 16;

    const long planeHW = (long)NH * NW;
    const float* x1b = x1 + (long)bb * NCH * planeHW;
    const float* x2b = x2 + (long)bb * NCH * planeHW;

    // ---- A fragments: x1[bb, :, h, w0+sub], 16-bit A 16x32 ISA layout ----
    // lane(half,sub): M = sub, element e=(2v+s): K = 2v+s + (v>=4 ? 8:0) + 8*half
    v16h fa0, fa1;
    {
        const float* p = x1b + (long)h * NW + (w0 + sub);
#pragma unroll
        for (int e = 0; e < 16; ++e) {
            const int v = e >> 1, s = e & 1;
            const int k = 2 * v + s + (v >= 4 ? 8 : 0) + 8 * half;
            fa0[e] = (_Float16)p[(long)k * planeHW];
            fa1[e] = (_Float16)p[(long)(k + 32) * planeHW];
        }
    }

    float* lo = lds_out[wave];

    for (int i = -SR; i <= SR; ++i) {
        const int  row   = h - i;
        const bool rowok = (row >= 0) && (row < NH);

        v8f acc0 = {};   // N tile 0: x2 cols w0-4 .. w0+11
        v8f acc1 = {};   // N tile 1: x2 cols w0+12 .. w0+27

#pragma unroll
        for (int t = 0; t < 2; ++t) {
            // B fragment: 16-bit B 32x16, column-striped per lane:
            // N = sub, element e: K = e + (e >= 8 ? 8 : 0) + 8*half
            const int  col   = w0 - 4 + 16 * t + sub;
            const bool colok = rowok && (col >= 0) && (col < NW);
            v16h fb0 = {}, fb1 = {};
            if (colok) {
                const float* q = x2b + (long)row * NW + col;
#pragma unroll
                for (int e = 0; e < 16; ++e) {
                    const int k = e + (e >= 8 ? 8 : 0) + 8 * half;
                    fb0[e] = (_Float16)q[(long)k * planeHW];
                    fb1[e] = (_Float16)q[(long)(k + 32) * planeHW];
                }
            }
            if (t == 0) {
                acc0 = __builtin_amdgcn_wmma_f32_16x16x32_f16(
                    false, fa0, false, fb0, (short)0, acc0, false, false);
                acc0 = __builtin_amdgcn_wmma_f32_16x16x32_f16(
                    false, fa1, false, fb1, (short)0, acc0, false, false);
            } else {
                acc1 = __builtin_amdgcn_wmma_f32_16x16x32_f16(
                    false, fa0, false, fb0, (short)0, acc1, false, false);
                acc1 = __builtin_amdgcn_wmma_f32_16x16x32_f16(
                    false, fa1, false, fb1, (short)0, acc1, false, false);
            }
        }

        // ---- extract diagonals: D layout lane -> (m = r + 8*half, n = sub) ----
        // tile t: shift j = m - n + 4 - 16*t, valid for -4..4; each (j,m) hit once
#pragma unroll
        for (int r = 0; r < 8; ++r) {
            const int m  = r + 8 * half;
            const int j0 = m - sub + 4;
            if (j0 >= -SR && j0 <= SR) lo[(j0 + SR) * 16 + m] = acc0[r];
            const int j1 = m - sub - 12;
            if (j1 >= -SR && j1 <= SR) lo[(j1 + SR) * 16 + m] = acc1[r];
        }

        __syncthreads();   // uniform across all 8 waves (i-loop is uniform)

        // ---- coalesced store: 9 k-planes x 16 contiguous floats ----
#pragma unroll
        for (int pass = 0; pass < 5; ++pass) {
            const int idx = pass * 32 + lane;
            if (idx < DD * 16) {
                const int jr = idx >> 4;           // 0..8  -> j = jr - 4
                const int m  = idx & 15;
                int kk = DD * i + (jr - SR);
                if (kk < 0) kk += NC;              // torch negative-index wrap
                out[(((long)bb * NC + kk) * NH + h) * NW + (w0 + m)] =
                    lo[idx] * (1.0f / NC);
            }
        }

        __syncthreads();
    }
}

extern "C" void kernel_launch(void* const* d_in, const int* in_sizes, int n_in,
                              void* d_out, int out_size, void* d_ws, size_t ws_size,
                              hipStream_t stream)
{
    const float* x1 = (const float*)d_in[0];
    const float* x2 = (const float*)d_in[1];
    float* out = (float*)d_out;

    const int wgroups = (NW / 16) / WPB;                 // 2
    const int nblocks = wgroups * NH * NB;               // 3072
    costvol_wmma_kernel<<<nblocks, 32 * WPB, 0, stream>>>(x1, x2, out);
}
